// BatchNeuralKB_20306605375522
// MI455X (gfx1250) — matrix operand order repaired and checked by reading, not compiled, not run on hardware
//
#include <hip/hip_runtime.h>
#include <hip/hip_bf16.h>
#include <stdint.h>

typedef __attribute__((ext_vector_type(16))) __bf16 v16bf;
typedef __attribute__((ext_vector_type(8)))  float  v8f;

constexpr int kE      = 128;   // per-array feature dim
constexpr int kD      = 384;   // 3*kE concat dim
constexpr int MTILE   = 64;    // queries per block
constexpr int NTILE   = 128;   // facts per staged tile
constexpr int NCHUNK  = 1024;  // facts per block
constexpr int KSTEP   = 32;    // bf16 WMMA K
constexpr int LSTRIDE = 392;   // padded bf16 row stride (384+8): conflict-free b128 frags

union Frag { v16bf v; uint4 q[2]; };

// fp32 -> packed 2x bf16, round-to-nearest-even
__device__ inline unsigned pack2bf(float a, float b) {
  unsigned ua = __float_as_uint(a), ub = __float_as_uint(b);
  ua += 0x7FFFu + ((ua >> 16) & 1u);
  ub += 0x7FFFu + ((ub >> 16) & 1u);
  return (ua >> 16) | (ub & 0xFFFF0000u);
}

// order-preserving float <-> uint key (ascending), for atomic min on uints
__device__ inline unsigned fkey(float x) {
  unsigned u = __float_as_uint(x);
  return (u & 0x80000000u) ? ~u : (u | 0x80000000u);
}
__device__ inline float fkey_dec(unsigned k) {
  return (k & 0x80000000u) ? __uint_as_float(k & 0x7FFFFFFFu)
                           : __uint_as_float(~k);
}

__global__ __launch_bounds__(256)
void init_minkey_kernel(unsigned* minkey, int n) {
  int i = blockIdx.x * blockDim.x + threadIdx.x;
  if (i < n) minkey[i] = 0xFFFFFFFFu;
}

// one wave per row: ||row||^2 over the 3 concatenated arrays
__global__ __launch_bounds__(256)
void fsq_kernel(const float* __restrict__ r, const float* __restrict__ a1,
                const float* __restrict__ a2, float* __restrict__ fsq) {
  int row  = (blockIdx.x * blockDim.x + threadIdx.x) >> 5;
  int lane = threadIdx.x & 31;
  const float4 x = ((const float4*)(r  + (size_t)row * kE))[lane];
  const float4 y = ((const float4*)(a1 + (size_t)row * kE))[lane];
  const float4 z = ((const float4*)(a2 + (size_t)row * kE))[lane];
  float s = x.x*x.x + x.y*x.y + x.z*x.z + x.w*x.w
          + y.x*y.x + y.y*y.y + y.z*y.z + y.w*y.w
          + z.x*z.x + z.y*z.y + z.z*z.z + z.w*z.w;
  #pragma unroll
  for (int off = 16; off; off >>= 1) s += __shfl_xor(s, off, 32);
  if (lane == 0) fsq[row] = s;
}

// finish: out[b] = exp(-max(0, ||q_b||^2 + min_f(||f||^2 - 2 q.f)) / 2)
__global__ __launch_bounds__(256)
void final_kernel(const float* __restrict__ r, const float* __restrict__ a1,
                  const float* __restrict__ a2, const unsigned* __restrict__ minkey,
                  float* __restrict__ out) {
  int row  = (blockIdx.x * blockDim.x + threadIdx.x) >> 5;
  int lane = threadIdx.x & 31;
  const float4 x = ((const float4*)(r  + (size_t)row * kE))[lane];
  const float4 y = ((const float4*)(a1 + (size_t)row * kE))[lane];
  const float4 z = ((const float4*)(a2 + (size_t)row * kE))[lane];
  float s = x.x*x.x + x.y*x.y + x.z*x.z + x.w*x.w
          + y.x*y.x + y.y*y.y + y.z*y.z + y.w*y.w
          + z.x*z.x + z.y*z.y + z.z*z.z + z.w*z.w;
  #pragma unroll
  for (int off = 16; off; off >>= 1) s += __shfl_xor(s, off, 32);
  if (lane == 0) {
    float d2 = fmaxf(s + fkey_dec(minkey[row]), 0.0f);
    out[row] = expf(-0.5f * d2);
  }
}

__global__ __launch_bounds__(256, 2)
void kb_wmma_kernel(const float* __restrict__ rel, const float* __restrict__ a1,
                    const float* __restrict__ a2, const float* __restrict__ fr,
                    const float* __restrict__ fa1, const float* __restrict__ fa2,
                    const float* __restrict__ fsq, unsigned* __restrict__ minkey) {
  extern __shared__ char smem[];
  uint16_t* s_q   = (uint16_t*)smem;                       // [MTILE][LSTRIDE] bf16
  uint16_t* s_f   = s_q + MTILE * LSTRIDE;                 // [NTILE][LSTRIDE] bf16
  unsigned* s_min = (unsigned*)(s_f + NTILE * LSTRIDE);    // [MTILE]

  const int tid  = threadIdx.x;
  const int lane = tid & 31;
  const int wave = tid >> 5;
  const int lm   = lane & 15;
  const int lh   = lane >> 4;
  const int wmb  = (wave >> 2) * 32;   // wave M base (2 wave-rows)
  const int wnb  = (wave & 3) * 32;    // wave N base (4 wave-cols)

  if (tid < MTILE) s_min[tid] = 0xFFFFFFFFu;

  // ---- stage Q tile: MTILE x 384 fp32 -> bf16 LDS ----
  const int qrow0 = blockIdx.y * MTILE;
  #pragma unroll
  for (int i = 0; i < (MTILE * (kD / 4)) / 256; ++i) {     // 24 iters
    int idx = i * 256 + tid;
    int row = idx / (kD / 4);
    int c4  = idx % (kD / 4);                              // float4 column
    int arr = c4 >> 5;                                     // which of 3 arrays
    int cc  = (c4 & 31) << 2;
    const float* src = (arr == 0) ? rel : (arr == 1 ? a1 : a2);
    float4 v = *(const float4*)(src + (size_t)(qrow0 + row) * kE + cc);
    *(uint2*)(s_q + row * LSTRIDE + c4 * 4) =
        make_uint2(pack2bf(v.x, v.y), pack2bf(v.z, v.w));
  }

  float rmin[2][8];
  #pragma unroll
  for (int mt = 0; mt < 2; ++mt)
    #pragma unroll
    for (int j = 0; j < 8; ++j) rmin[mt][j] = 3.4e38f;

  const int chunk0 = blockIdx.x * NCHUNK;
  for (int t = 0; t < NCHUNK / NTILE; ++t) {
    const int nb = chunk0 + t * NTILE;
    __syncthreads();   // s_f readers from prev iter done; also fences s_q stores (t==0)

    // ---- stage facts tile: NTILE x 384 fp32 -> bf16 LDS ----
    #pragma unroll
    for (int i = 0; i < (NTILE * (kD / 4)) / 256; ++i) {   // 48 iters
      int idx = i * 256 + tid;
      int row = idx / (kD / 4);
      int c4  = idx % (kD / 4);
      int arr = c4 >> 5;
      int cc  = (c4 & 31) << 2;
      const float* src = (arr == 0) ? fr : (arr == 1 ? fa1 : fa2);
      float4 v = *(const float4*)(src + (size_t)(nb + row) * kE + cc);
      *(uint2*)(s_f + row * LSTRIDE + c4 * 4) =
          make_uint2(pack2bf(v.x, v.y), pack2bf(v.z, v.w));
    }
    __syncthreads();

    v8f acc[2][2] = {};
    #pragma unroll
    for (int ks = 0; ks < kD / KSTEP; ++ks) {              // 12 K-steps
      const int k0 = ks * KSTEP;
      Frag A[2], B[2];
      #pragma unroll
      for (int mt = 0; mt < 2; ++mt) {
        // A frag: lanes 0-15 row m, K {0..7,16..23}; lanes 16-31 K {8..15,24..31}
        const uint16_t* p = s_q + (wmb + mt * 16 + lm) * LSTRIDE + k0 + lh * 8;
        A[mt].q[0] = *(const uint4*)p;
        A[mt].q[1] = *(const uint4*)(p + 16);
      }
      #pragma unroll
      for (int nt = 0; nt < 2; ++nt) {
        // B frag: lanes 0-15 col n, K {0..7,16..23}; lanes 16-31 K {8..15,24..31}
        const uint16_t* p = s_f + (wnb + nt * 16 + lm) * LSTRIDE + k0 + lh * 8;
        B[nt].q[0] = *(const uint4*)p;
        B[nt].q[1] = *(const uint4*)(p + 16);
      }
      #pragma unroll
      for (int mt = 0; mt < 2; ++mt)
        #pragma unroll
        for (int nt = 0; nt < 2; ++nt)
          acc[mt][nt] = __builtin_amdgcn_wmma_f32_16x16x32_bf16(
              false, A[mt].v, false, B[nt].v, (short)0, acc[mt][nt], false, false);
    }

    // fold: running min over n of (||f_n||^2 - 2 * q.f)
    #pragma unroll
    for (int nt = 0; nt < 2; ++nt) {
      float fs = fsq[nb + wnb + nt * 16 + lm];
      #pragma unroll
      for (int mt = 0; mt < 2; ++mt)
        #pragma unroll
        for (int j = 0; j < 8; ++j)
          rmin[mt][j] = fminf(rmin[mt][j], fs - 2.0f * acc[mt][nt][j]);
    }
  }

  // reduce min across the 16 lanes of each half-wave, then LDS, then global
  #pragma unroll
  for (int mt = 0; mt < 2; ++mt)
    #pragma unroll
    for (int j = 0; j < 8; ++j) {
      float v = rmin[mt][j];
      v = fminf(v, __shfl_xor(v, 1, 32));
      v = fminf(v, __shfl_xor(v, 2, 32));
      v = fminf(v, __shfl_xor(v, 4, 32));
      v = fminf(v, __shfl_xor(v, 8, 32));
      if (lm == 0) {
        int mloc = wmb + mt * 16 + lh * 8 + j;   // C layout: lanes16-31 are m+8
        atomicMin(&s_min[mloc], fkey(v));
      }
    }
  __syncthreads();
  if (tid < MTILE) atomicMin(&minkey[qrow0 + tid], s_min[tid]);
}

extern "C" void kernel_launch(void* const* d_in, const int* in_sizes, int n_in,
                              void* d_out, int out_size, void* d_ws, size_t ws_size,
                              hipStream_t stream) {
  const float* rel  = (const float*)d_in[0];
  const float* arg1 = (const float*)d_in[1];
  const float* arg2 = (const float*)d_in[2];
  const float* fr   = (const float*)d_in[3];
  const float* fa1  = (const float*)d_in[4];
  const float* fa2  = (const float*)d_in[5];
  float* out = (float*)d_out;

  const int B = in_sizes[0] / kE;   // 2048
  const int F = in_sizes[3] / kE;   // 65536

  float*    fsq    = (float*)d_ws;
  unsigned* minkey = (unsigned*)((char*)d_ws + (size_t)F * sizeof(float));

  init_minkey_kernel<<<(B + 255) / 256, 256, 0, stream>>>(minkey, B);
  fsq_kernel<<<F / 8, 256, 0, stream>>>(fr, fa1, fa2, fsq);

  const size_t lds = (size_t)(MTILE + NTILE) * LSTRIDE * sizeof(uint16_t)
                   + MTILE * sizeof(unsigned);
  hipFuncSetAttribute((const void*)kb_wmma_kernel,
                      hipFuncAttributeMaxDynamicSharedMemorySize, (int)lds);
  dim3 grid(F / NCHUNK, B / MTILE);
  kb_wmma_kernel<<<grid, 256, lds, stream>>>(rel, arg1, arg2, fr, fa1, fa2,
                                             fsq, minkey);

  final_kernel<<<B / 8, 256, 0, stream>>>(rel, arg1, arg2, minkey, out);
}
